// SpatialGNNFixed_10642928959597
// MI455X (gfx1250) — matrix operand order repaired and checked by reading, not compiled, not run on hardware
//
#include <hip/hip_runtime.h>
#include <hip/hip_bf16.h>
#include <stdint.h>

#define NNODES 50000
#define NEDGES 800000
#define IN_DIM 128
#define HID 64
#define HEADS 4
#define EDGE_DIM 8
#define NLAYERS 2
#define HC (HEADS * HID) /* 256 */
#define NEG_SLOPE 0.2f

typedef __attribute__((ext_vector_type(16))) __bf16 v16bf;
typedef __attribute__((ext_vector_type(8)))  __bf16 v8bf;
typedef __attribute__((ext_vector_type(8)))  float  v8f;

// float atomic max via signed/unsigned bit-pattern trick (init = 0xFF bytes)
static __device__ __forceinline__ void atomicMaxF(float* addr, float v) {
  if (v >= 0.0f) atomicMax((int*)addr, __float_as_int(v));
  else           atomicMin((unsigned int*)addr, __float_as_uint(v));
}

// ---------------------------------------------------------------------------
// Bulk f32 -> bf16 conversion (8 elements / thread, native packed cvt).
// ---------------------------------------------------------------------------
__global__ void cvt_bf16_kernel(const float* __restrict__ in,
                                __bf16* __restrict__ out, int n8) {
  int i = blockIdx.x * blockDim.x + threadIdx.x;
  if (i >= n8) return;
  const float4* p = (const float4*)in + (size_t)i * 2;
  float4 a = p[0], b = p[1];
  v8bf o;
  o[0] = (__bf16)a.x; o[1] = (__bf16)a.y; o[2] = (__bf16)a.z; o[3] = (__bf16)a.w;
  o[4] = (__bf16)b.x; o[5] = (__bf16)b.y; o[6] = (__bf16)b.z; o[7] = (__bf16)b.w;
  *((v8bf*)out + i) = o;
}

// ---------------------------------------------------------------------------
// Pack B[K,Nc] (f32 row-major) into per-lane WMMA bf16 fragment order:
// frag (tile_n, kb): lane holds 16 contiguous bf16 = B[kb*32 + half*16 + i][tile_n*16 + l16]
// ---------------------------------------------------------------------------
__global__ void pack_B_kernel(const float* __restrict__ B,
                              __bf16* __restrict__ Bp, int K, int Nc) {
  int w = (int)((blockIdx.x * blockDim.x + threadIdx.x) >> 5);
  int nkb = K >> 5, ntile = Nc >> 4;
  if (w >= ntile * nkb) return;
  int tile_n = w / nkb, kb = w % nkb;
  int lane = threadIdx.x & 31, half = lane >> 4, l16 = lane & 15;
  const float* Bc = B + (size_t)(kb * 32 + half * 16) * Nc + tile_n * 16 + l16;
  __bf16* o = Bp + ((size_t)w * 32 + lane) * 16;
#pragma unroll
  for (int i = 0; i < 16; ++i) o[i] = (__bf16)Bc[(size_t)i * Nc];
}

// ---------------------------------------------------------------------------
// C[M,NC] = A[M,K]@B[K,NC] (+bias), A bf16 row-major, B pre-packed fragments.
// One wave computes a 16x64 strip: A frag loaded once per K-block (2x b128),
// reused across 4 WMMA tiles. K/NC compile-time => all fragment addresses are
// base + constant immediate offsets. C/D layout: col=l16, row = v + half*8.
// ---------------------------------------------------------------------------
template <int K, int NC>
__global__ void gemm_bf16_wmma(const __bf16* __restrict__ A,
                               const __bf16* __restrict__ Bp,
                               const float* __restrict__ bias,
                               float* __restrict__ C, int M) {
  constexpr int NKB = K / 32;
  constexpr int NT4 = NC / 64;
  int wave = (int)((blockIdx.x * blockDim.x + threadIdx.x) >> 5);
  int T = (M >> 4) * NT4;
  if (wave >= T) return;
  int g_n = wave % NT4;
  int tile_m = wave / NT4;
  int lane = threadIdx.x & 31, half = lane >> 4, l16 = lane & 15;

  v8f acc[4] = {};
  const __bf16* Arow  = A + (size_t)(tile_m * 16 + l16) * K;
  const __bf16* Bbase = Bp + ((size_t)(g_n * 4 * NKB) * 32 + lane) * 16;

#pragma unroll
  for (int kb = 0; kb < NKB; ++kb) {
    v8bf lo = *(const v8bf*)(Arow + kb * 32 + half * 16);
    v8bf hi = *(const v8bf*)(Arow + kb * 32 + half * 16 + 8);
    v16bf a = __builtin_shufflevector(lo, hi, 0, 1, 2, 3, 4, 5, 6, 7,
                                      8, 9, 10, 11, 12, 13, 14, 15);
#pragma unroll
    for (int t = 0; t < 4; ++t) {
      v16bf b = *(const v16bf*)(Bbase + (size_t)(t * NKB + kb) * 32 * 16);
      acc[t] = __builtin_amdgcn_wmma_f32_16x16x32_bf16(false, a, false, b,
                                                       (short)0, acc[t], false, false);
    }
  }

  float* Crow = C + (size_t)(tile_m * 16 + half * 8) * NC + g_n * 64 + l16;
#pragma unroll
  for (int t = 0; t < 4; ++t) {
    float bv = bias ? bias[g_n * 64 + t * 16 + l16] : 0.0f;
#pragma unroll
    for (int v = 0; v < 8; ++v) Crow[(size_t)v * NC + t * 16] = acc[t][v] + bv;
  }
}

// ---------------------------------------------------------------------------
__global__ void attr_count_kernel(const int* __restrict__ attr, int* __restrict__ cnt) {
  int e = blockIdx.x * blockDim.x + threadIdx.x;
  if (e < NEDGES) atomicAdd(&cnt[attr[e]], 1);
}

// ee4[c][j] = edge_emb[c] @ We (c=0..2), ee4[3] = mean-edge-feature @ We
__global__ void ee4_kernel(const float* __restrict__ eemb,
                           const float* __restrict__ We,
                           const int* __restrict__ cnt,
                           float* __restrict__ ee4) {
  int j = threadIdx.x; // 0..255
#pragma unroll
  for (int c = 0; c < 3; ++c) {
    float s = 0.f;
#pragma unroll
    for (int d = 0; d < EDGE_DIM; ++d) s += eemb[c * EDGE_DIM + d] * We[d * HC + j];
    ee4[c * HC + j] = s;
  }
  float s = 0.f;
  const float invE = 1.0f / (float)NEDGES;
#pragma unroll
  for (int d = 0; d < EDGE_DIM; ++d) {
    float m = ((float)cnt[0] * eemb[d] + (float)cnt[1] * eemb[EDGE_DIM + d] +
               (float)cnt[2] * eemb[2 * EDGE_DIM + d]) * invE;
    s += m * We[d * HC + j];
  }
  ee4[3 * HC + j] = s;
}

// ---------------------------------------------------------------------------
// Pass 1: per-edge attention logits + segment max over dst. 1 wave / edge.
// Lane handles 8 contiguous channels; lanes [8h..8h+7] cover head h exactly.
__global__ void edge_score_kernel(const float* __restrict__ xl,
                                  const float* __restrict__ xr,
                                  const float* __restrict__ ee4,
                                  const int* __restrict__ src,
                                  const int* __restrict__ dst,
                                  const int* __restrict__ attr,
                                  const float* __restrict__ att,
                                  float* __restrict__ sbuf,
                                  float* __restrict__ smax,
                                  int ET) {
  int e = (int)((blockIdx.x * blockDim.x + threadIdx.x) >> 5);
  if (e >= ET) return;
  int lane = threadIdx.x & 31;
  int sN, dN, cs;
  if (e < NEDGES) { sN = src[e]; dN = dst[e]; cs = attr[e]; }
  else            { sN = dN = e - NEDGES; cs = 3; }
  int j = lane * 8;
  const float4* pl = (const float4*)(xl + (size_t)sN * HC + j);
  const float4* pr = (const float4*)(xr + (size_t)dN * HC + j);
  const float4* pe = (const float4*)(ee4 + cs * HC + j);
  const float4* pa = (const float4*)(att + j);
  float part = 0.0f;
#pragma unroll
  for (int t = 0; t < 2; ++t) {
    float4 a = pl[t], b = pr[t], c = pe[t], w = pa[t];
    float m;
    m = a.x + b.x + c.x; part += (m > 0.f ? m : NEG_SLOPE * m) * w.x;
    m = a.y + b.y + c.y; part += (m > 0.f ? m : NEG_SLOPE * m) * w.y;
    m = a.z + b.z + c.z; part += (m > 0.f ? m : NEG_SLOPE * m) * w.z;
    m = a.w + b.w + c.w; part += (m > 0.f ? m : NEG_SLOPE * m) * w.w;
  }
  part += __shfl_down(part, 4, 8);
  part += __shfl_down(part, 2, 8);
  part += __shfl_down(part, 1, 8);
  if ((lane & 7) == 0) {
    int h = lane >> 3;
    sbuf[(size_t)e * HEADS + h] = part;
    atomicMaxF(&smax[(size_t)dN * HEADS + h], part);
  }
}

// Pass 2: a = exp(s - smax[dst]); denom[dst] += a; outacc[dst] += a * xl[src]
__global__ void edge_accum_kernel(const float* __restrict__ xl,
                                  const float* __restrict__ sbuf,
                                  const float* __restrict__ smax,
                                  const int* __restrict__ src,
                                  const int* __restrict__ dst,
                                  float* __restrict__ denom,
                                  float* __restrict__ outacc,
                                  int ET) {
  int e = (int)((blockIdx.x * blockDim.x + threadIdx.x) >> 5);
  if (e >= ET) return;
  int lane = threadIdx.x & 31;
  int sN, dN;
  if (e < NEDGES) { sN = src[e]; dN = dst[e]; }
  else            { sN = dN = e - NEDGES; }
  int h = lane >> 3;
  float a = __expf(sbuf[(size_t)e * HEADS + h] - smax[(size_t)dN * HEADS + h]);
  if ((lane & 7) == 0) atomicAdd(&denom[(size_t)dN * HEADS + h], a);
  int j = lane * 8;
  const float4* pl = (const float4*)(xl + (size_t)sN * HC + j);
  float* po = outacc + (size_t)dN * HC + j;
  float4 v0 = pl[0], v1 = pl[1];
  atomicAdd(po + 0, a * v0.x); atomicAdd(po + 1, a * v0.y);
  atomicAdd(po + 2, a * v0.z); atomicAdd(po + 3, a * v0.w);
  atomicAdd(po + 4, a * v1.x); atomicAdd(po + 5, a * v1.y);
  atomicAdd(po + 6, a * v1.z); atomicAdd(po + 7, a * v1.w);
}

// head-mean + bias + residual + LayerNorm. 1 wave / node, lane = channels {c, c+32}.
__global__ void node_finalize_kernel(const float* __restrict__ outacc,
                                     const float* __restrict__ denom,
                                     const float* __restrict__ bias,
                                     const float* __restrict__ hin,
                                     const float* __restrict__ lnw,
                                     const float* __restrict__ lnb,
                                     float* __restrict__ hout, int Nn) {
  int n = (int)((blockIdx.x * blockDim.x + threadIdx.x) >> 5);
  if (n >= Nn) return;
  int lane = threadIdx.x & 31;
  float inv[HEADS];
#pragma unroll
  for (int h = 0; h < HEADS; ++h) inv[h] = 1.0f / denom[(size_t)n * HEADS + h];
  float t0 = 0.f, t1 = 0.f;
#pragma unroll
  for (int h = 0; h < HEADS; ++h) {
    t0 += outacc[(size_t)n * HC + h * HID + lane]      * inv[h];
    t1 += outacc[(size_t)n * HC + h * HID + lane + 32] * inv[h];
  }
  t0 = t0 * 0.25f + bias[lane]      + hin[(size_t)n * HID + lane];
  t1 = t1 * 0.25f + bias[lane + 32] + hin[(size_t)n * HID + lane + 32];
  float s = t0 + t1;
#pragma unroll
  for (int off = 16; off >= 1; off >>= 1) s += __shfl_xor(s, off, 32);
  float mu = s * (1.0f / 64.0f);
  float u0 = t0 - mu, u1 = t1 - mu;
  float q = u0 * u0 + u1 * u1;
#pragma unroll
  for (int off = 16; off >= 1; off >>= 1) q += __shfl_xor(q, off, 32);
  float r = rsqrtf(q * (1.0f / 64.0f) + 1e-5f);
  hout[(size_t)n * HID + lane]      = u0 * r * lnw[lane]      + lnb[lane];
  hout[(size_t)n * HID + lane + 32] = u1 * r * lnw[lane + 32] + lnb[lane + 32];
}

// ---------------------------------------------------------------------------
extern "C" void kernel_launch(void* const* d_in, const int* in_sizes, int n_in,
                              void* d_out, int out_size, void* d_ws, size_t ws_size,
                              hipStream_t stream) {
  const float* x      = (const float*)d_in[0];
  const int*   eidx   = (const int*)d_in[1];
  const int*   eattr  = (const int*)d_in[2];
  const float* proj_w = (const float*)d_in[3];
  const float* proj_b = (const float*)d_in[4];
  const float* eemb   = (const float*)d_in[5];
  const float* Wl     = (const float*)d_in[6];
  const float* Wr     = (const float*)d_in[7];
  const float* We     = (const float*)d_in[8];
  const float* att    = (const float*)d_in[9];
  const float* bias   = (const float*)d_in[10];
  const float* ln_w   = (const float*)d_in[11];
  const float* ln_b   = (const float*)d_in[12];
  const int* srcv = eidx;
  const int* dstv = eidx + NEDGES;

  char* base = (char*)d_ws;
  float* h0   = (float*)base; base += (size_t)NNODES * HID * 4;
  float* h1   = (float*)base; base += (size_t)NNODES * HID * 4;
  float* xl   = (float*)base; base += (size_t)NNODES * HC * 4;
  float* xr   = (float*)base; base += (size_t)NNODES * HC * 4; // -> outacc in pass 2
  float* sbuf = (float*)base; base += (size_t)(NEDGES + NNODES) * HEADS * 4;
  float* smax = (float*)base; base += (size_t)NNODES * HEADS * 4;
  float* den  = (float*)base; base += (size_t)NNODES * HEADS * 4;
  float* ee4  = (float*)base; base += 4 * HC * 4;
  int*   cnt  = (int*)base;   base += 16;
  __bf16* xbf  = (__bf16*)base; base += (size_t)NNODES * IN_DIM * 2;
  __bf16* hbf  = (__bf16*)base; base += (size_t)NNODES * HID * 2;
  __bf16* pw_p = (__bf16*)base; base += (size_t)IN_DIM * HID * 2;
  __bf16* wl_p = (__bf16*)base; base += (size_t)HID * HC * 2;
  __bf16* wr_p = (__bf16*)base; base += (size_t)HID * HC * 2;

  const int ET = NEDGES + NNODES;

  hipMemsetAsync(cnt, 0, 3 * sizeof(int), stream);
  attr_count_kernel<<<(NEDGES + 255) / 256, 256, 0, stream>>>(eattr, cnt);

  // input projection: h0 = x @ proj_w + proj_b
  {
    int n8 = NNODES * IN_DIM / 8;
    cvt_bf16_kernel<<<(n8 + 255) / 256, 256, 0, stream>>>(x, xbf, n8);
    int wp = (IN_DIM / 32) * (HID / 16);
    pack_B_kernel<<<(wp + 7) / 8, 256, 0, stream>>>(proj_w, pw_p, IN_DIM, HID);
    int T = (NNODES / 16) * (HID / 64);
    gemm_bf16_wmma<IN_DIM, HID><<<(T + 7) / 8, 256, 0, stream>>>(xbf, pw_p, proj_b,
                                                                 h0, NNODES);
  }

  const float* hin = h0;
  for (int l = 0; l < NLAYERS; ++l) {
    float* hout = (l == NLAYERS - 1) ? (float*)d_out : h1;

    int n8 = NNODES * HID / 8;
    cvt_bf16_kernel<<<(n8 + 255) / 256, 256, 0, stream>>>(hin, hbf, n8);
    int wp = (HID / 32) * (HC / 16);
    pack_B_kernel<<<(wp + 7) / 8, 256, 0, stream>>>(Wl + (size_t)l * HID * HC, wl_p, HID, HC);
    pack_B_kernel<<<(wp + 7) / 8, 256, 0, stream>>>(Wr + (size_t)l * HID * HC, wr_p, HID, HC);

    int T = (NNODES / 16) * (HC / 64);
    gemm_bf16_wmma<HID, HC><<<(T + 7) / 8, 256, 0, stream>>>(hbf, wl_p, nullptr, xl, NNODES);
    gemm_bf16_wmma<HID, HC><<<(T + 7) / 8, 256, 0, stream>>>(hbf, wr_p, nullptr, xr, NNODES);
    ee4_kernel<<<1, HC, 0, stream>>>(eemb, We + (size_t)l * EDGE_DIM * HC, cnt, ee4);

    hipMemsetAsync(smax, 0xFF, (size_t)NNODES * HEADS * sizeof(float), stream);
    edge_score_kernel<<<(ET + 7) / 8, 256, 0, stream>>>(
        xl, xr, ee4, srcv, dstv, eattr, att + (size_t)l * HC, sbuf, smax, ET);

    hipMemsetAsync(xr, 0, (size_t)NNODES * HC * sizeof(float), stream); // -> outacc
    hipMemsetAsync(den, 0, (size_t)NNODES * HEADS * sizeof(float), stream);
    edge_accum_kernel<<<(ET + 7) / 8, 256, 0, stream>>>(
        xl, sbuf, smax, srcv, dstv, den, xr, ET);

    node_finalize_kernel<<<(NNODES + 7) / 8, 256, 0, stream>>>(
        xr, den, bias + (size_t)l * HID, hin,
        ln_w + (size_t)l * HID, ln_b + (size_t)l * HID, hout, NNODES);
    hin = hout;
  }
  (void)in_sizes; (void)n_in; (void)out_size; (void)ws_size;
}
